// GRU_13091060318761
// MI455X (gfx1250) — matrix operand order repaired and checked by reading, not compiled, not run on hardware
//
#include <hip/hip_runtime.h>
#include <hip/hip_bf16.h>

// ---------------------------------------------------------------------------
// 2-layer GRU for MI455X (gfx1250, wave32, WMMA).
//   B=32, T=512, D=H=1024, gates=3H=3072.
// Phase 1: bf16 conversion of inputs/weights.
// Phase 2: WMMA GEMM  xp = in_bf16 @ W_ih^T + b_ih   (fp32 out)
// Phase 3: persistent recurrence kernel, W_hh slice staged in 96KB LDS
//          (CDNA5 WGP has 320KB), v_wmma_f32_16x16x32_bf16 inner product,
//          software grid barrier per timestep.
// ---------------------------------------------------------------------------

typedef __attribute__((ext_vector_type(16))) __bf16 v16bf;
typedef __attribute__((ext_vector_type(8)))  __bf16 v8bf;
typedef __attribute__((ext_vector_type(8)))  float  v8f;

#define B_DIM 32
#define T_DIM 512
#define D_DIM 1024
#define H_DIM 1024
#define G3H   3072

__device__ __forceinline__ unsigned short f2bf(float f) {
    unsigned u = __float_as_uint(f);
    unsigned r = (u + 0x7FFFu + ((u >> 16) & 1u)) >> 16;
    return (unsigned short)r;
}

// ---------------------------------------------------------------- conversions
__global__ void __launch_bounds__(256) conv_f32_to_bf16(
    const float* __restrict__ in, unsigned short* __restrict__ out, int n) {
    int i = blockIdx.x * 256 + threadIdx.x;
    if (i < n) out[i] = f2bf(in[i]);
}

__global__ void __launch_bounds__(256) init_hidden(
    const float* __restrict__ hin, float* __restrict__ h32,
    unsigned short* __restrict__ hbf, int n) {
    int i = blockIdx.x * 256 + threadIdx.x;
    if (i < n) { float v = hin[i]; h32[i] = v; hbf[i] = f2bf(v); }
}

// ------------------------------------------------------------- projection GEMM
// xp[M,3072] = in_bf[M,1024] @ w_bf[3072,1024]^T + bias
// block: 256 threads = 8 waves; each wave computes a 16x64 tile (4 WMMA accums);
// block tile = 128(M) x 64(N).
__global__ void __launch_bounds__(256) gemm_proj(
    const unsigned short* __restrict__ inbf,
    const unsigned short* __restrict__ wbf,
    const float* __restrict__ bias,
    float* __restrict__ xp, int M, int K) {
    const int wave = threadIdx.x >> 5;
    const int lane = threadIdx.x & 31;
    const int m0 = blockIdx.x * 128 + wave * 16;
    const int n0 = blockIdx.y * 64;
    const int lhalf = (lane < 16) ? 0 : 8;      // unused; A also loads at bhalf
    const int bhalf = (lane < 16) ? 0 : 16;     // B-fragment K half
    const size_t arow = (size_t)(m0 + (lane & 15)) * K;

    v8f acc[4] = {};
    for (int k0 = 0; k0 < K; k0 += 32) {
        v16bf a;
        a = *(const v16bf*)(inbf + arow + k0 + bhalf);
#pragma unroll
        for (int nt = 0; nt < 4; nt++) {
            int ncol = n0 + nt * 16 + (lane & 15);
            v16bf b = *(const v16bf*)(wbf + (size_t)ncol * K + k0 + bhalf);
            acc[nt] = __builtin_amdgcn_wmma_f32_16x16x32_bf16(
                false, a, false, b, (short)0, acc[nt], false, false);
        }
    }
#pragma unroll
    for (int nt = 0; nt < 4; nt++) {
        int ncol = n0 + nt * 16 + (lane & 15);
        float bv = bias[ncol];
#pragma unroll
        for (int v = 0; v < 8; v++) {
            int mrow = m0 + v + ((lane >= 16) ? 8 : 0);
            xp[(size_t)mrow * G3H + ncol] = acc[nt][v] + bv;
        }
    }
}

// ----------------------------------------------------------------- recurrence
// 64 persistent blocks x 192 threads (6 waves). Block owns 16 H-columns.
// LDS: 48 rows x 1024 bf16 W_hh slice (96KB) + 3x32x16 fp32 gate buffer.
__device__ __forceinline__ void grid_sync(unsigned* bar, unsigned expected) {
    __threadfence();
    __syncthreads();
    if (threadIdx.x == 0) {
        __hip_atomic_fetch_add(bar, 1u, __ATOMIC_RELEASE, __HIP_MEMORY_SCOPE_AGENT);
        while (__hip_atomic_load(bar, __ATOMIC_ACQUIRE, __HIP_MEMORY_SCOPE_AGENT)
               < expected) {
            __builtin_amdgcn_s_sleep(4);
        }
    }
    __syncthreads();
    __threadfence();
}

__global__ void __launch_bounds__(192) gru_recur(
    const float* __restrict__ xp,            // [B,T,3H], b_ih folded in
    const unsigned short* __restrict__ whh,  // [3H,H] bf16
    const float* __restrict__ bhh,           // [3H]
    float* h32,                              // [B,H] fp32 state
    unsigned short* hbf,                     // [B,H] bf16 state
    unsigned short* ybf,                     // [B,T,H] bf16 out (or null)
    float* y32,                              // [B,T,H] fp32 out (or null)
    float* __restrict__ h_final,             // [B,H] slot in d_out
    unsigned* bar, int T) {
    extern __shared__ unsigned short wlds[];            // 48*1024 bf16
    float* hp_s = (float*)(wlds + 48 * 1024);           // [3][32][16]

    const int tid = threadIdx.x;
    const int wave = tid >> 5;
    const int lane = tid & 31;
    const int colbase = blockIdx.x * 16;

    // stage W_hh slice: global rows g*H + colbase+n  ->  LDS row g*16+n
    for (int idx = tid; idx < 48 * 128; idx += 192) {
        int r = idx >> 7;
        int c = (idx & 127) * 8;
        int g = r >> 4, n = r & 15;
        *(uint4*)(wlds + r * H_DIM + c) =
            *(const uint4*)(whh + (size_t)(g * H_DIM + colbase + n) * H_DIM + c);
    }
    __syncthreads();

    const int mt = wave & 1;        // batch tile 0/1
    const int g  = wave >> 1;       // gate 0..2 (waves 0..5)
    const int lhalf = (lane < 16) ? 0 : 8;
    const int bhalf = (lane < 16) ? 0 : 16;
    float bv = 0.f;
    if (wave < 6) bv = bhh[g * H_DIM + colbase + (lane & 15)];

    for (int t = 0; t < T; t++) {
        if (wave < 6) {
            const int arow = mt * 16 + (lane & 15);
            const unsigned short* hrow = hbf + (size_t)arow * H_DIM;
            const unsigned short* wrow =
                wlds + (size_t)(g * 16 + (lane & 15)) * H_DIM + bhalf;
            if (t + 1 < T)   // warm L2/WGP$ for next step's xp rows
                __builtin_prefetch(
                    xp + ((size_t)arow * T + t + 1) * G3H + g * H_DIM + colbase, 0, 3);
            v8f acc = {};
            for (int k0 = 0; k0 < H_DIM; k0 += 32) {
                v16bf a;
                a = *(const v16bf*)(hrow + k0 + bhalf);
                v16bf b = *(const v16bf*)(wrow + k0);
                acc = __builtin_amdgcn_wmma_f32_16x16x32_bf16(
                    false, a, false, b, (short)0, acc, false, false);
            }
#pragma unroll
            for (int v = 0; v < 8; v++) {
                int m = mt * 16 + v + ((lane >= 16) ? 8 : 0);
                hp_s[(g * 32 + m) * 16 + (lane & 15)] = acc[v] + bv;
            }
        }
        __syncthreads();

        for (int idx = tid; idx < 512; idx += 192) {
            int m = idx >> 4, n = idx & 15;
            int col = colbase + n;
            size_t xrow = ((size_t)m * T + t) * G3H;
            float pr = xp[xrow + col]         + hp_s[(0 * 32 + m) * 16 + n];
            float pz = xp[xrow + 1024 + col]  + hp_s[(1 * 32 + m) * 16 + n];
            float hpn =                         hp_s[(2 * 32 + m) * 16 + n];
            float r = 1.f / (1.f + __expf(-pr));
            float z = 1.f / (1.f + __expf(-pz));
            float nn = tanhf(xp[xrow + 2048 + col] + r * hpn);
            float hprev = h32[m * H_DIM + col];
            float hnew = (1.f - z) * nn + z * hprev;
            h32[m * H_DIM + col] = hnew;
            hbf[m * H_DIM + col] = f2bf(hnew);
            size_t yidx = ((size_t)m * T + t) * H_DIM + col;
            if (ybf) ybf[yidx] = f2bf(hnew);
            if (y32) y32[yidx] = hnew;
            if (t == T - 1) h_final[m * H_DIM + col] = hnew;
        }
        grid_sync(bar, (unsigned)gridDim.x * (unsigned)(t + 1));
    }
}

// ---------------------------------------------------------------------- host
extern "C" void kernel_launch(void* const* d_in, const int* in_sizes, int n_in,
                              void* d_out, int out_size, void* d_ws, size_t ws_size,
                              hipStream_t stream) {
    const float* x      = (const float*)d_in[0];
    const float* hidden = (const float*)d_in[1];
    const float* w_ih0  = (const float*)d_in[2];
    const float* w_hh0  = (const float*)d_in[3];
    const float* b_ih0  = (const float*)d_in[4];
    const float* b_hh0  = (const float*)d_in[5];
    const float* w_ih1  = (const float*)d_in[6];
    const float* w_hh1  = (const float*)d_in[7];
    const float* b_ih1  = (const float*)d_in[8];
    const float* b_hh1  = (const float*)d_in[9];
    float* out = (float*)d_out;                 // [B,T,H] y1  ++  [2,B,H] h

    char* ws = (char*)d_ws;
    size_t off = 0;
    auto alloc = [&](size_t bytes) -> void* {
        off = (off + 255) & ~(size_t)255;
        void* p = ws + off;
        off += bytes;
        return p;
    };
    unsigned*       bar   = (unsigned*)alloc(256);
    unsigned short* xbf   = (unsigned short*)alloc((size_t)B_DIM * T_DIM * D_DIM * 2); // also y0 bf16
    unsigned short* wihb0 = (unsigned short*)alloc((size_t)G3H * D_DIM * 2);
    unsigned short* whhb0 = (unsigned short*)alloc((size_t)G3H * H_DIM * 2);
    unsigned short* wihb1 = (unsigned short*)alloc((size_t)G3H * H_DIM * 2);
    unsigned short* whhb1 = (unsigned short*)alloc((size_t)G3H * H_DIM * 2);
    float*          xp    = (float*)alloc((size_t)B_DIM * T_DIM * G3H * 4);
    float*          h32   = (float*)alloc((size_t)B_DIM * H_DIM * 4);
    unsigned short* hbf   = (unsigned short*)alloc((size_t)B_DIM * H_DIM * 2);

    const int M = B_DIM * T_DIM;                      // 16384
    const int nX = M * D_DIM;                         // 16.7M
    const int nW = G3H * H_DIM;                       // 3.1M
    const size_t yElems = (size_t)B_DIM * T_DIM * H_DIM;
    float* hOut0 = out + yElems;
    float* hOut1 = out + yElems + (size_t)B_DIM * H_DIM;

    // ---- bf16 conversions
    conv_f32_to_bf16<<<(nX + 255) / 256, 256, 0, stream>>>(x, xbf, nX);
    conv_f32_to_bf16<<<(nW + 255) / 256, 256, 0, stream>>>(w_ih0, wihb0, nW);
    conv_f32_to_bf16<<<(nW + 255) / 256, 256, 0, stream>>>(w_hh0, whhb0, nW);
    conv_f32_to_bf16<<<(nW + 255) / 256, 256, 0, stream>>>(w_ih1, wihb1, nW);
    conv_f32_to_bf16<<<(nW + 255) / 256, 256, 0, stream>>>(w_hh1, whhb1, nW);

    dim3 gGemm(M / 128, G3H / 64);
    const size_t recurLds = 48 * 1024 * 2 + 3 * 32 * 16 * 4;   // 96KB + 6KB
    const int nH = B_DIM * H_DIM;

    // ---- layer 0
    init_hidden<<<(nH + 255) / 256, 256, 0, stream>>>(hidden, h32, hbf, nH);
    gemm_proj<<<gGemm, 256, 0, stream>>>(xbf, wihb0, b_ih0, xp, M, D_DIM);
    (void)hipMemsetAsync(bar, 0, 256, stream);
    gru_recur<<<64, 192, recurLds, stream>>>(xp, whhb0, b_hh0, h32, hbf,
                                             /*ybf=*/xbf, /*y32=*/nullptr,
                                             hOut0, bar, T_DIM);

    // ---- layer 1 (input = y0 bf16, reusing xbf buffer)
    init_hidden<<<(nH + 255) / 256, 256, 0, stream>>>(hidden + nH, h32, hbf, nH);
    gemm_proj<<<gGemm, 256, 0, stream>>>(xbf, wihb1, b_ih1, xp, M, H_DIM);
    (void)hipMemsetAsync(bar, 0, 256, stream);
    gru_recur<<<64, 192, recurLds, stream>>>(xp, whhb1, b_hh1, h32, hbf,
                                             /*ybf=*/nullptr, /*y32=*/out,
                                             hOut1, bar, T_DIM);
}
